// TransformerLayerShard_85383949844717
// MI455X (gfx1250) — compile-verified, matrix operand order checked by reading, not run on hardware
//
#include <hip/hip_runtime.h>

// ---------------------------------------------------------------------------
// Types / helpers
// ---------------------------------------------------------------------------
typedef unsigned int   u32;
typedef unsigned short u16;
typedef __attribute__((ext_vector_type(4)))  u32    u32x4;
typedef __attribute__((ext_vector_type(16))) __bf16 v16bf;
typedef __attribute__((ext_vector_type(8)))  float  v8f;

union Frag { u32x4 u[2]; v16bf v; };   // 16 bf16 = 8 VGPRs, two b128 chunks

__device__ __forceinline__ u16 f2bf(float f) {
    u32 u = __float_as_uint(f);
    u32 r = u + 0x7fffu + ((u >> 16) & 1u);   // round-to-nearest-even
    return (u16)(r >> 16);
}
__device__ __forceinline__ float b2f(u16 h) {
    return __uint_as_float(((u32)h) << 16);
}
__device__ __forceinline__ float gelu_f(float x) {
    // tanh approximation (jax.nn.gelu default)
    return 0.5f * x * (1.f + tanhf(0.7978845608028654f * (x + 0.044715f * x * x * x)));
}

#define SEQL   2048
#define DMODEL 2048
#define NH     16
#define DHD    128
#define DFFN   8192
#define BNECK  1088   // BACKGROUND_LEN + STYLE_VECS_LEN
#define BGL    1024

// ---------------------------------------------------------------------------
// f32 -> bf16 cast (weights, grid-stride)
// ---------------------------------------------------------------------------
__global__ void cast_bf16_kernel(const float* __restrict__ src,
                                 u16* __restrict__ dst, int n) {
    int i = blockIdx.x * blockDim.x + threadIdx.x;
    int stride = gridDim.x * blockDim.x;
    for (; i < n; i += stride) dst[i] = f2bf(src[i]);
}

// ---------------------------------------------------------------------------
// LayerNorm (one row per block) -> bf16
// ---------------------------------------------------------------------------
__global__ __launch_bounds__(256) void layernorm_cast_kernel(
    const float* __restrict__ x, const float* __restrict__ scale,
    const float* __restrict__ offset, u16* __restrict__ xn) {
    __shared__ float red[256];
    const int row = blockIdx.x, tid = threadIdx.x;
    float vals[8];
    float s = 0.f, sq = 0.f;
#pragma unroll
    for (int i = 0; i < 8; ++i) {
        float v = x[(size_t)row * DMODEL + tid + i * 256];
        vals[i] = v; s += v; sq += v * v;
    }
    red[tid] = s; __syncthreads();
    for (int o = 128; o > 0; o >>= 1) { if (tid < o) red[tid] += red[tid + o]; __syncthreads(); }
    float mean = red[0] * (1.f / DMODEL);
    __syncthreads();
    red[tid] = sq; __syncthreads();
    for (int o = 128; o > 0; o >>= 1) { if (tid < o) red[tid] += red[tid + o]; __syncthreads(); }
    float var  = red[0] * (1.f / DMODEL) - mean * mean;
    float rstd = rsqrtf(var + 1e-5f);
#pragma unroll
    for (int i = 0; i < 8; ++i) {
        int c = tid + i * 256;
        float y = scale[c] * (vals[i] - mean) * rstd + offset[c];
        xn[(size_t)row * DMODEL + c] = f2bf(y);
    }
}

// ---------------------------------------------------------------------------
// RoPE, in place on bf16 q,k. q additionally scaled by 1/sqrt(DH).
// ---------------------------------------------------------------------------
__global__ void rope_kernel(u16* __restrict__ q, u16* __restrict__ k) {
    int idx = blockIdx.x * blockDim.x + threadIdx.x;   // SEQ*NH*(DHD/2) threads
    int i   = idx & 63;            // rotary pair index
    int rem = idx >> 6;
    int h   = rem & (NH - 1);
    int pos = rem >> 4;
    float inv = __powf(10000.f, -2.f * (float)i * (1.f / DHD));
    float ang = (float)pos * inv;
    float sn, cs; __sincosf(ang, &sn, &cs);
    size_t base = ((size_t)pos * NH + h) * DHD + 2 * i;
    const float qs = 0.08838834764831845f;  // 1/sqrt(128)
    float q0 = b2f(q[base]), q1 = b2f(q[base + 1]);
    q[base]     = f2bf((q0 * cs - q1 * sn) * qs);
    q[base + 1] = f2bf((q1 * cs + q0 * sn) * qs);
    float k0 = b2f(k[base]), k1 = b2f(k[base + 1]);
    k[base]     = f2bf(k0 * cs - k1 * sn);
    k[base + 1] = f2bf(k1 * cs + k0 * sn);
}

// ---------------------------------------------------------------------------
// WMMA GEMM: C[M,N] = A[M,K](bf16) x B[K,N](bf16), f32 accumulate.
// Block tile 128x256, 8 waves (2x4), each wave 64x64 (4x4 WMMA accums), BK=32.
// Register-staged double-buffered LDS; one barrier per K-step; L2 prefetch of
// the K-step after next (global_prefetch_b8).
// flags: bit0 = GELU, bit1 = add existing Cf. bias added if non-null.
// ---------------------------------------------------------------------------
__global__ __launch_bounds__(256) void wmma_gemm_kernel(
    const u16* __restrict__ A, const u16* __restrict__ B,
    const float* __restrict__ bias, float* __restrict__ Cf,
    u16* __restrict__ Cb, int M, int N, int K, int flags) {
    __shared__ u32 sA[2][128 * 16];     // A tile [128][32] bf16 (dwords), x2
    __shared__ u16 sB[2][256 * 32];     // B tile transposed [n][k], x2

    const int tid  = threadIdx.x;
    const int wave = tid >> 5, lane = tid & 31;
    const int lm = lane & 15, hi = lane >> 4;
    const int wm = wave >> 2, wn = wave & 3;          // 2 x 4 wave grid
    const int mBase = blockIdx.y * 128, nBase = blockIdx.x * 256;

    v8f zero = 0.f;
    v8f acc[4][4];
#pragma unroll
    for (int r = 0; r < 4; ++r)
#pragma unroll
        for (int c = 0; c < 4; ++c) acc[r][c] = zero;

    const u32* A32 = (const u32*)A;
    const u32* B32 = (const u32*)B;
    const int ldaD = K >> 1, ldbD = N >> 1;

    u32 ra[8], rb[16];

    auto loadA = [&](int kb) {
#pragma unroll
        for (int i = 0; i < 8; ++i) {            // 128 rows x 16 dwords
            int idx = tid + i * 256;
            int r = idx >> 4, c = idx & 15;
            ra[i] = A32[(size_t)(mBase + r) * ldaD + (kb >> 1) + c];
        }
    };
    auto loadB = [&](int kb) {
#pragma unroll
        for (int i = 0; i < 16; ++i) {           // 32 k-rows x 128 dwords
            int idx = tid + i * 256;
            int k = idx >> 7, nd = idx & 127;
            rb[i] = B32[(size_t)(kb + k) * ldbD + (nBase >> 1) + nd];
        }
    };
    auto storeA = [&](int buf) {
#pragma unroll
        for (int i = 0; i < 8; ++i) {
            int idx = tid + i * 256;
            int r = idx >> 4, c = idx & 15;
            sA[buf][r * 16 + c] = ra[i];
        }
    };
    auto storeB = [&](int buf) {
#pragma unroll
        for (int i = 0; i < 16; ++i) {
            int idx = tid + i * 256;
            int k = idx >> 7, nd = idx & 127;
            u32 d = rb[i];
            sB[buf][(2 * nd) * 32 + k]     = (u16)(d & 0xffffu);
            sB[buf][(2 * nd + 1) * 32 + k] = (u16)(d >> 16);
        }
    };

    // prologue: stage K-step 0
    loadA(0); loadB(0);
    storeA(0); storeB(0);
    __syncthreads();

    const int nsteps = K >> 5;
    for (int s = 0; s < nsteps; ++s) {
        const int cur = s & 1;
        const bool more = (s + 1) < nsteps;
        if (more) {
            loadA((s + 1) * 32);
            loadB((s + 1) * 32);
            if ((s + 2) < nsteps) {   // L2 prefetch of K-step after next
                __builtin_prefetch(&A32[(size_t)(mBase + (tid >> 4)) * ldaD + ((s + 2) * 32 >> 1) + (tid & 15)], 0, 3);
                __builtin_prefetch(&B32[(size_t)((s + 2) * 32 + (tid >> 7)) * ldbD + (nBase >> 1) + (tid & 127)], 0, 3);
            }
        }

        // fragments + 16 WMMA
        Frag fa[4], fb[4];
        const u32* pa = sA[cur];
        const u32* pb = (const u32*)sB[cur];
#pragma unroll
        for (int r = 0; r < 4; ++r) {
            int row = wm * 64 + r * 16 + lm;
            fa[r].u[0] = *(const u32x4*)&pa[row * 16 + 4 * hi];
            fa[r].u[1] = *(const u32x4*)&pa[row * 16 + 8 + 4 * hi];
        }
#pragma unroll
        for (int c = 0; c < 4; ++c) {
            int col = wn * 64 + c * 16 + lm;
            fb[c].u[0] = *(const u32x4*)&pb[col * 16 + 4 * hi];
            fb[c].u[1] = *(const u32x4*)&pb[col * 16 + 8 + 4 * hi];
        }
#pragma unroll
        for (int r = 0; r < 4; ++r)
#pragma unroll
            for (int c = 0; c < 4; ++c)
                acc[r][c] = __builtin_amdgcn_wmma_f32_16x16x32_bf16(
                    false, fa[r].v, false, fb[c].v, (short)0, acc[r][c], false, false);

        if (more) { storeA(cur ^ 1); storeB(cur ^ 1); }
        __syncthreads();
    }

    // epilogue
#pragma unroll
    for (int r = 0; r < 4; ++r)
#pragma unroll
        for (int c = 0; c < 4; ++c)
#pragma unroll
            for (int v = 0; v < 8; ++v) {
                int row = mBase + wm * 64 + r * 16 + v + 8 * hi;
                int col = nBase + wn * 64 + c * 16 + lm;
                float val = acc[r][c][v];
                if (bias) val += bias[col];
                if (flags & 1) val = gelu_f(val);
                size_t off = (size_t)row * N + col;
                if (Cf) { if (flags & 2) val += Cf[off]; Cf[off] = val; }
                if (Cb) Cb[off] = f2bf(val);
            }
}

// ---------------------------------------------------------------------------
// Flash attention: block = (q-tile of 64 rows, head); 4 waves x 16 q rows.
// Streams 32-key tiles; S = Q K^T via 4 WMMA (dh chunks), online softmax,
// O += P V via 8 WMMA. Waves skip fully-masked tiles (wave-uniform branch,
// EXEC stays all-ones for WMMA). q pre-scaled by 1/sqrt(DH) in rope_kernel.
// ---------------------------------------------------------------------------
__global__ __launch_bounds__(128) void flash_attn_kernel(
    const u16* __restrict__ qg, const u16* __restrict__ kg,
    const u16* __restrict__ vg, const float* __restrict__ attn_bias,
    u16* __restrict__ outv) {
    __shared__ u16 Kt[32 * DHD];        // [key][dh]
    __shared__ u16 Vt[DHD * 32];        // [dh][key] (transposed)
    __shared__ u16 Pb[4 * 16 * 32];     // per-wave P buffer

    const int tid  = threadIdx.x;
    const int wave = tid >> 5, lane = tid & 31;
    const int lm = lane & 15, hi = lane >> 4;
    const int h     = blockIdx.y;
    const int qtile = blockIdx.x;
    const int qbase = qtile * 64 + wave * 16;

    // preload Q fragments: 4 dh-chunks of 32
    Frag aq[4];
    {
        const u32* q32  = (const u32*)qg;
        const u32* qrow = q32 + ((size_t)(qbase + lm) * NH + h) * (DHD / 2);
#pragma unroll
        for (int c = 0; c < 4; ++c) {
            aq[c].u[0] = *(const u32x4*)(qrow + c * 16 + 4 * hi);
            aq[c].u[1] = *(const u32x4*)(qrow + c * 16 + 8 + 4 * hi);
        }
    }

    v8f zero = 0.f;
    v8f acc[8];
#pragma unroll
    for (int nc = 0; nc < 8; ++nc) acc[nc] = zero;
    float mrow[8], lrow[8];
#pragma unroll
    for (int v = 0; v < 8; ++v) { mrow[v] = -1e30f; lrow[v] = 0.f; }

    const u32* k32 = (const u32*)kg;
    const u32* v32 = (const u32*)vg;
    u32* Kt32 = (u32*)Kt;
    const int nkt = qtile * 2 + 2;      // causal: key tiles up to q block end

    for (int kt = 0; kt < nkt; ++kt) {
        const int kbase = kt * 32;
        // stage K tile [32 keys][128 dh]
#pragma unroll
        for (int i = 0; i < 16; ++i) {
            int idx = tid + i * 128;     // 0..2047 dwords
            int key = idx >> 6, c = idx & 63;
            Kt32[key * 64 + c] = k32[((size_t)(kbase + key) * NH + h) * 64 + c];
        }
        // stage V transposed [128 dh][32 keys]
#pragma unroll
        for (int i = 0; i < 16; ++i) {
            int idx = tid + i * 128;
            int key = idx >> 6, c = idx & 63;   // c = dh pair index
            u32 d = v32[((size_t)(kbase + key) * NH + h) * 64 + c];
            Vt[(2 * c) * 32 + key]     = (u16)(d & 0xffffu);
            Vt[(2 * c + 1) * 32 + key] = (u16)(d >> 16);
        }
        // L2 prefetch next K/V tiles
        if (kt + 1 < nkt) {
            __builtin_prefetch(&k32[((size_t)(kbase + 32 + (tid >> 6)) * NH + h) * 64 + (tid & 63)], 0, 3);
            __builtin_prefetch(&v32[((size_t)(kbase + 32 + (tid >> 6)) * NH + h) * 64 + (tid & 63)], 0, 3);
        }
        __syncthreads();

        // wave-uniform skip of fully-masked tiles (causal)
        if (kbase <= qbase + 15) {
            // S tiles (16q x 16k) x2, accumulate over 4 dh chunks
            v8f s0 = zero, s1 = zero;
            const u32* KtU = (const u32*)Kt;
#pragma unroll
            for (int c = 0; c < 4; ++c) {
                Frag b0, b1;
                b0.u[0] = *(const u32x4*)(KtU + lm * 64 + c * 16 + 4 * hi);
                b0.u[1] = *(const u32x4*)(KtU + lm * 64 + c * 16 + 8 + 4 * hi);
                b1.u[0] = *(const u32x4*)(KtU + (16 + lm) * 64 + c * 16 + 4 * hi);
                b1.u[1] = *(const u32x4*)(KtU + (16 + lm) * 64 + c * 16 + 8 + 4 * hi);
                s0 = __builtin_amdgcn_wmma_f32_16x16x32_bf16(false, aq[c].v, false, b0.v, (short)0, s0, false, false);
                s1 = __builtin_amdgcn_wmma_f32_16x16x32_bf16(false, aq[c].v, false, b1.v, (short)0, s1, false, false);
            }

            // bias + mask + online softmax; write P (bf16) to per-wave LDS
            u16* Pw = Pb + wave * 512;
            const int key0 = kbase + lm, key1 = kbase + 16 + lm;
#pragma unroll
            for (int v = 0; v < 8; ++v) {
                int q = qbase + v + 8 * hi;
                float msk0 = ((key0 <= q) && !((q >= BNECK) && (key0 < BGL))) ? 0.f : -1e10f;
                float msk1 = ((key1 <= q) && !((q >= BNECK) && (key1 < BGL))) ? 0.f : -1e10f;
                float f0 = s0[v] + attn_bias[(size_t)q * SEQL + key0] + msk0;
                float f1 = s1[v] + attn_bias[(size_t)q * SEQL + key1] + msk1;
                float t = fmaxf(f0, f1);
                t = fmaxf(t, __shfl_xor(t, 1, 32));
                t = fmaxf(t, __shfl_xor(t, 2, 32));
                t = fmaxf(t, __shfl_xor(t, 4, 32));
                t = fmaxf(t, __shfl_xor(t, 8, 32));
                float mn = fmaxf(mrow[v], t);
                float alpha = __expf(mrow[v] - mn);
                float p0 = __expf(f0 - mn), p1 = __expf(f1 - mn);
                float rs = p0 + p1;
                rs += __shfl_xor(rs, 1, 32);
                rs += __shfl_xor(rs, 2, 32);
                rs += __shfl_xor(rs, 4, 32);
                rs += __shfl_xor(rs, 8, 32);
                lrow[v] = lrow[v] * alpha + rs;
                mrow[v] = mn;
#pragma unroll
                for (int nc = 0; nc < 8; ++nc) acc[nc][v] *= alpha;
                Pw[(v + 8 * hi) * 32 + lm]      = f2bf(p0);
                Pw[(v + 8 * hi) * 32 + 16 + lm] = f2bf(p1);
            }
            // per-wave LDS is hardware-ordered (DS in-order); stop compiler
            // from reordering the typed reads above the stores.
            asm volatile("" ::: "memory");

            // O += P x V
            Frag pa;
            const u32* PwU = (const u32*)Pw;
            pa.u[0] = *(const u32x4*)(PwU + lm * 16 + 4 * hi);
            pa.u[1] = *(const u32x4*)(PwU + lm * 16 + 8 + 4 * hi);
            const u32* VtU = (const u32*)Vt;
#pragma unroll
            for (int nc = 0; nc < 8; ++nc) {
                Frag bv;
                bv.u[0] = *(const u32x4*)(VtU + (nc * 16 + lm) * 16 + 4 * hi);
                bv.u[1] = *(const u32x4*)(VtU + (nc * 16 + lm) * 16 + 8 + 4 * hi);
                acc[nc] = __builtin_amdgcn_wmma_f32_16x16x32_bf16(false, pa.v, false, bv.v, (short)0, acc[nc], false, false);
            }
        }
        __syncthreads();   // before next tile overwrites Kt/Vt
    }

    // write attn_vec (bf16) for the Wo GEMM
#pragma unroll
    for (int nc = 0; nc < 8; ++nc)
#pragma unroll
        for (int v = 0; v < 8; ++v) {
            int q  = qbase + v + 8 * hi;
            int dh = nc * 16 + lm;
            float val = acc[nc][v] / lrow[v];
            outv[(size_t)q * DMODEL + h * DHD + dh] = f2bf(val);
        }
}

// ---------------------------------------------------------------------------
// Launcher
// ---------------------------------------------------------------------------
extern "C" void kernel_launch(void* const* d_in, const int* in_sizes, int n_in,
                              void* d_out, int out_size, void* d_ws, size_t ws_size,
                              hipStream_t stream) {
    (void)in_sizes; (void)n_in; (void)out_size; (void)ws_size;
    const float* x         = (const float*)d_in[0];
    const float* attn_bias = (const float*)d_in[1];
    const float* ln_scale  = (const float*)d_in[2];
    const float* ln_offset = (const float*)d_in[3];
    const float* Wq = (const float*)d_in[4];
    const float* Wk = (const float*)d_in[5];
    const float* Wv = (const float*)d_in[6];
    const float* Wo = (const float*)d_in[7];
    const float* W1 = (const float*)d_in[8];
    const float* b1 = (const float*)d_in[9];
    const float* W2 = (const float*)d_in[10];
    const float* b2 = (const float*)d_in[11];
    float* out = (float*)d_out;

    char* ws = (char*)d_ws;
    const size_t MB = 1024 * 1024;
    u16* xn  = (u16*)(ws + 0 * MB);     // 8 MB
    u16* Wqb = (u16*)(ws + 8 * MB);     // 8 MB
    u16* Wkb = (u16*)(ws + 16 * MB);
    u16* Wvb = (u16*)(ws + 24 * MB);
    u16* Wob = (u16*)(ws + 32 * MB);
    u16* W1b = (u16*)(ws + 40 * MB);    // 32 MB
    u16* W2b = (u16*)(ws + 72 * MB);    // 32 MB
    u16* qb  = (u16*)(ws + 104 * MB);   // 8 MB
    u16* kbf = (u16*)(ws + 112 * MB);
    u16* vbf = (u16*)(ws + 120 * MB);
    u16* av  = (u16*)(ws + 128 * MB);   // 8 MB
    u16* hid = (u16*)(ws + 136 * MB);   // 32 MB  (total 168 MB)

    // 1. cast weights to bf16
    cast_bf16_kernel<<<2048, 256, 0, stream>>>(Wq, Wqb, DMODEL * DMODEL);
    cast_bf16_kernel<<<2048, 256, 0, stream>>>(Wk, Wkb, DMODEL * DMODEL);
    cast_bf16_kernel<<<2048, 256, 0, stream>>>(Wv, Wvb, DMODEL * DMODEL);
    cast_bf16_kernel<<<2048, 256, 0, stream>>>(Wo, Wob, DMODEL * DMODEL);
    cast_bf16_kernel<<<4096, 256, 0, stream>>>(W1, W1b, DMODEL * DFFN);
    cast_bf16_kernel<<<4096, 256, 0, stream>>>(W2, W2b, DFFN * DMODEL);

    // 2. layernorm
    layernorm_cast_kernel<<<SEQL, 256, 0, stream>>>(x, ln_scale, ln_offset, xn);

    // 3. q,k,v projections (bf16 out); block tile 128x256
    dim3 gD(DMODEL / 256, SEQL / 128);
    wmma_gemm_kernel<<<gD, 256, 0, stream>>>(xn, Wqb, nullptr, nullptr, qb, SEQL, DMODEL, DMODEL, 0);
    wmma_gemm_kernel<<<gD, 256, 0, stream>>>(xn, Wkb, nullptr, nullptr, kbf, SEQL, DMODEL, DMODEL, 0);
    wmma_gemm_kernel<<<gD, 256, 0, stream>>>(xn, Wvb, nullptr, nullptr, vbf, SEQL, DMODEL, DMODEL, 0);

    // 4. RoPE (in place; q scaled by 1/sqrt(DH))
    rope_kernel<<<(SEQL * NH * (DHD / 2)) / 256, 256, 0, stream>>>(qb, kbf);

    // 5. flash attention
    flash_attn_kernel<<<dim3(SEQL / 64, NH), 128, 0, stream>>>(qb, kbf, vbf, attn_bias, av);

    // 6. attn_out = attn_vec @ Wo  (f32 -> d_out)
    wmma_gemm_kernel<<<gD, 256, 0, stream>>>(av, Wob, nullptr, out, nullptr, SEQL, DMODEL, DMODEL, 0);

    // 7. hidden = gelu(xn @ W1 + b1)  (bf16)
    dim3 gF(DFFN / 256, SEQL / 128);
    wmma_gemm_kernel<<<gF, 256, 0, stream>>>(xn, W1b, b1, nullptr, hid, SEQL, DFFN, DMODEL, 1);

    // 8. d_out += hidden @ W2 + b2
    wmma_gemm_kernel<<<gD, 256, 0, stream>>>(hid, W2b, b2, out, nullptr, SEQL, DMODEL, DFFN, 2);
}
